// FGCN_73796128079920
// MI455X (gfx1250) — matrix operand order repaired and checked by reading, not compiled, not run on hardware
//
#include <hip/hip_runtime.h>

typedef __attribute__((ext_vector_type(16))) __bf16 v16bf;
typedef __attribute__((ext_vector_type(8)))  float  v8f;

#define TB 256
#define BM 128          // block tile M (8 waves x 16 rows)
#define BN 64           // block tile N (4 wmma tiles per wave)
#define KT 32
#define ROW_U32 17      // 16 data uints (32 bf16) + 1 pad per LDS row

__device__ __forceinline__ unsigned int f2bf(float f) {
  unsigned int u = __float_as_uint(f);
  u += 0x7FFFu + ((u >> 16) & 1u);        // round-to-nearest-even
  return u >> 16;
}
__device__ __forceinline__ unsigned int pk2(float a, float b) {
  return f2bf(a) | (f2bf(b) << 16);
}

// C[M,N] = act( A[M,K] @ B[K,N] + bias ), fp32 in/out, bf16 WMMA inside.
// A is split at column `ksplit`: k <  ksplit -> A0 (row stride ksplit)
//                                k >= ksplit -> A1 (row stride K-ksplit)
// (pass A1==A0, ksplit==K for a plain GEMM; split mode implements concat)
// Each of the 8 waves computes a 16x64 strip: 4 accumulators sharing 1 A-frag.
// All 4 B fragments are loaded into distinct registers BEFORE the 4 WMMAs so
// the compiler does not need WAR-hazard v_nops between wmma and ds_load.
__global__ __launch_bounds__(TB) void gemm_bf16_wmma(
    const float* __restrict__ A0, const float* __restrict__ A1, int ksplit,
    const float* __restrict__ B, const float* __restrict__ bias,
    float* __restrict__ C, int M, int N, int K, int act)
{
  __shared__ unsigned int As[BM * ROW_U32];   // BM x KT bf16, row-major (M,K)
  __shared__ unsigned int Bs[BN * ROW_U32];   // BN x KT bf16, transposed (N,K)

  const int tid  = threadIdx.x;
  const int wave = tid >> 5;      // 0..7 -> M strip
  const int lane = tid & 31;
  const int half = lane >> 4;
  const int l15  = lane & 15;

  const int blockM = blockIdx.x * BM;
  const int blockN = blockIdx.y * BN;

  const int lda0 = ksplit;
  const int lda1 = K - ksplit;

  // A staging: 2 threads per row, 16 floats each (4 float4 -> 8 packed uints)
  const int am = tid >> 1;            // 0..127
  const int ak = (tid & 1) * 16;      // 0 or 16
  int gm = blockM + am;
  if (gm >= M) gm = M - 1;            // clamp loads; stores are guarded

  // B staging: 8 threads per K-row, 8 floats each
  const int bk = tid >> 3;            // 0..31
  const int bn = (tid & 7) * 8;       // 0,8,...,56

  v8f acc[4] = {{}, {}, {}, {}};

  for (int k0 = 0; k0 < K; k0 += KT) {
    { // ---- stage A tile (BM x KT) ----
      const int kg = k0 + ak;
      const float* ap = (kg < ksplit) ? (A0 + (size_t)gm * lda0 + kg)
                                      : (A1 + (size_t)gm * lda1 + (kg - ksplit));
      float4 f0 = *reinterpret_cast<const float4*>(ap);
      float4 f1 = *reinterpret_cast<const float4*>(ap + 4);
      float4 f2 = *reinterpret_cast<const float4*>(ap + 8);
      float4 f3 = *reinterpret_cast<const float4*>(ap + 12);
      unsigned int* da = &As[am * ROW_U32 + (ak >> 1)];
      da[0] = pk2(f0.x, f0.y);
      da[1] = pk2(f0.z, f0.w);
      da[2] = pk2(f1.x, f1.y);
      da[3] = pk2(f1.z, f1.w);
      da[4] = pk2(f2.x, f2.y);
      da[5] = pk2(f2.z, f2.w);
      da[6] = pk2(f3.x, f3.y);
      da[7] = pk2(f3.z, f3.w);
    }
    { // ---- stage B tile transposed (BN cols x KT rows) ----
      const float* bp = B + (size_t)(k0 + bk) * N + blockN + bn;
      float4 b0 = *reinterpret_cast<const float4*>(bp);
      float4 b1 = *reinterpret_cast<const float4*>(bp + 4);
      unsigned short* Bs16 = reinterpret_cast<unsigned short*>(Bs);
      Bs16[(bn + 0) * (2 * ROW_U32) + bk] = (unsigned short)f2bf(b0.x);
      Bs16[(bn + 1) * (2 * ROW_U32) + bk] = (unsigned short)f2bf(b0.y);
      Bs16[(bn + 2) * (2 * ROW_U32) + bk] = (unsigned short)f2bf(b0.z);
      Bs16[(bn + 3) * (2 * ROW_U32) + bk] = (unsigned short)f2bf(b0.w);
      Bs16[(bn + 4) * (2 * ROW_U32) + bk] = (unsigned short)f2bf(b1.x);
      Bs16[(bn + 5) * (2 * ROW_U32) + bk] = (unsigned short)f2bf(b1.y);
      Bs16[(bn + 6) * (2 * ROW_U32) + bk] = (unsigned short)f2bf(b1.z);
      Bs16[(bn + 7) * (2 * ROW_U32) + bk] = (unsigned short)f2bf(b1.w);
    }
    __syncthreads();

    // ---- load 1 A fragment + all 4 B fragments, then 4 WMMAs back-to-back ----
    union frag_t { v16bf v; unsigned int u[8]; };
    frag_t fa, fb0, fb1, fb2, fb3;
    {
      const unsigned int* arow = &As[(wave * 16 + l15) * ROW_U32];
#pragma unroll
      for (int j = 0; j < 8; ++j) {
        // A: lanes 0-15 hold K {0..7,16..23}, lanes 16-31 hold K {8..15,24..31}
        fa.u[j] = arow[j + (j & 4) + 4 * half];
      }
      const unsigned int* brow = &Bs[l15 * ROW_U32 + 4 * half];
#pragma unroll
      for (int j = 0; j < 8; ++j) {
        // B: lanes 0-15 hold K {0..7,16..23}, lanes 16-31 hold K {8..15,24..31} (col = l15)
        fb0.u[j] = brow[j + (j & 4)];
        fb1.u[j] = brow[16 * ROW_U32 + j + (j & 4)];
        fb2.u[j] = brow[32 * ROW_U32 + j + (j & 4)];
        fb3.u[j] = brow[48 * ROW_U32 + j + (j & 4)];
      }
    }
    acc[0] = __builtin_amdgcn_wmma_f32_16x16x32_bf16(
        false, fa.v, false, fb0.v, (short)0, acc[0], false, false);
    acc[1] = __builtin_amdgcn_wmma_f32_16x16x32_bf16(
        false, fa.v, false, fb1.v, (short)0, acc[1], false, false);
    acc[2] = __builtin_amdgcn_wmma_f32_16x16x32_bf16(
        false, fa.v, false, fb2.v, (short)0, acc[2], false, false);
    acc[3] = __builtin_amdgcn_wmma_f32_16x16x32_bf16(
        false, fa.v, false, fb3.v, (short)0, acc[3], false, false);
    __syncthreads();
  }

  // ---- epilogue: D layout -> global, fused bias/ReLU ----
  const int mbase = blockM + wave * 16 + 8 * half;     // rows mbase + r
  float* cp = C + (size_t)mbase * N + blockN + l15;
#pragma unroll
  for (int nt = 0; nt < 4; ++nt) {
    const float bv = bias ? bias[blockN + nt * 16 + l15] : 0.0f;
#pragma unroll
    for (int r = 0; r < 8; ++r) {
      if (mbase + r < M) {
        float v = acc[nt][r] + bv;
        if (act) v = fmaxf(v, 0.0f);
        cp[(size_t)r * N + nt * 16] = v;
      }
    }
  }
}

// Y[dst[e], :] += w[e] * X[src[e], :]  — scatter via fp32 global atomics.
// grid.x covers edges, grid.y covers F/8 column chunks (8 floats per thread).
__global__ __launch_bounds__(TB) void spmm_scatter(
    const int* __restrict__ dst, const int* __restrict__ src,
    const float* __restrict__ w, const float* __restrict__ X,
    float* __restrict__ Y, int E, int F)
{
  const int e = blockIdx.x * TB + threadIdx.x;
  if (e >= E) return;
  const int c = blockIdx.y * 8;
  const int s = src[e];
  const int d = dst[e];
  const float wt = w[e];
  const float* xp = X + (size_t)s * F + c;
  float4 x0 = *reinterpret_cast<const float4*>(xp);
  float4 x1 = *reinterpret_cast<const float4*>(xp + 4);
  float* yp = Y + (size_t)d * F + c;
  atomicAdd(yp + 0, wt * x0.x);
  atomicAdd(yp + 1, wt * x0.y);
  atomicAdd(yp + 2, wt * x0.z);
  atomicAdd(yp + 3, wt * x0.w);
  atomicAdd(yp + 4, wt * x1.x);
  atomicAdd(yp + 5, wt * x1.y);
  atomicAdd(yp + 6, wt * x1.z);
  atomicAdd(yp + 7, wt * x1.w);
}

// Y += bias (broadcast over rows), optional ReLU. F is a power of two.
__global__ __launch_bounds__(TB) void bias_act(
    float* __restrict__ Y, const float* __restrict__ b,
    long long total, int fmask, int relu)
{
  const long long t = (long long)blockIdx.x * TB + threadIdx.x;
  if (t >= total) return;
  float v = Y[t] + b[(int)(t & fmask)];
  if (relu) v = fmaxf(v, 0.0f);
  Y[t] = v;
}

extern "C" void kernel_launch(void* const* d_in, const int* in_sizes, int n_in,
                              void* d_out, int out_size, void* d_ws, size_t ws_size,
                              hipStream_t stream)
{
  (void)n_in; (void)out_size; (void)ws_size;

  const float* xd   = (const float*)d_in[0];
  const int*   eid  = (const int*)  d_in[1];
  const float* ewd  = (const float*)d_in[2];
  const int*   eifd = (const int*)  d_in[3];
  const float* ewfd = (const float*)d_in[4];
  const float* xs   = (const float*)d_in[5];
  const int*   eis  = (const int*)  d_in[6];
  const float* ews  = (const float*)d_in[7];
  const int*   eifs = (const int*)  d_in[8];
  const float* ewfs = (const float*)d_in[9];
  const float* W1d = (const float*)d_in[10]; const float* b1d = (const float*)d_in[11];
  const float* W2d = (const float*)d_in[12]; const float* b2d = (const float*)d_in[13];
  const float* W1s = (const float*)d_in[14]; const float* b1s = (const float*)d_in[15];
  const float* W2s = (const float*)d_in[16]; const float* b2s = (const float*)d_in[17];
  const float* Wfd = (const float*)d_in[18]; const float* bfd = (const float*)d_in[19];
  const float* Wfs = (const float*)d_in[20]; const float* bfs = (const float*)d_in[21];

  const int H1 = in_sizes[11];              // 256
  const int H2 = in_sizes[13];              // 128
  const int Fd = in_sizes[10] / H1;         // 256
  const int Fs = in_sizes[14] / H1;         // 256
  const int Nd = in_sizes[0] / Fd;          // 100000
  const int Ns = in_sizes[5] / Fs;          // 50000
  const int Ed = in_sizes[2];               // 3.2M
  const int Es = in_sizes[7];               // 1.6M

  // workspace: XW1 (Nd*H1) | H (Nd*H1) | HW2 (Nd*H2)  — reused for disease
  float* XW1  = (float*)d_ws;
  float* Hbuf = XW1  + (size_t)Nd * H1;
  float* HW2  = Hbuf + (size_t)Nd * H1;

  // output layout (flat, return order)
  float* emb1      = (float*)d_out;
  float* emb2      = emb1      + (size_t)Nd * H2;
  float* emb1_sim  = emb2      + (size_t)Ns * H2;
  float* emb1_feat = emb1_sim  + (size_t)Nd * H2;
  float* emb2_sim  = emb1_feat + (size_t)Nd * H2;
  float* emb2_feat = emb2_sim  + (size_t)Ns * H2;

  auto gemm = [&](const float* A0, const float* A1, int ksplit, const float* B,
                  const float* bias, float* C, int M, int N, int K, int act) {
    dim3 grid((M + BM - 1) / BM, N / BN);
    gemm_bf16_wmma<<<grid, TB, 0, stream>>>(A0, A1, ksplit, B, bias, C, M, N, K, act);
  };
  auto spmm = [&](const int* ei, const float* w, const float* X, float* Y,
                  int E, int Nn, int F) {
    hipMemsetAsync(Y, 0, (size_t)Nn * F * sizeof(float), stream);
    dim3 grid((E + TB - 1) / TB, F / 8);
    // reference: src = ei[1], dst = ei[0]  (rows of the (2,E) index array)
    spmm_scatter<<<grid, TB, 0, stream>>>(ei, ei + E, w, X, Y, E, F);
  };
  auto badd = [&](float* Y, const float* b, int Nn, int F, int relu) {
    long long total = (long long)Nn * F;
    unsigned blocks = (unsigned)((total + TB - 1) / TB);
    bias_act<<<dim3(blocks), TB, 0, stream>>>(Y, b, total, F - 1, relu);
  };
  auto gcn_tail = [&](const int* ei, const float* w, int E, int Nn,
                      const float* b1, const float* W2, const float* b2, float* emb) {
    spmm(ei, w, XW1, Hbuf, E, Nn, H1);          // H = adj @ (x@W1)
    badd(Hbuf, b1, Nn, H1, 1);                  // H = relu(H + b1)
    gemm(Hbuf, Hbuf, H1, W2, nullptr, HW2, Nn, H2, H1, 0);  // HW2 = H @ W2
    spmm(ei, w, HW2, emb, E, Nn, H2);           // emb = adj @ HW2
    badd(emb, b2, Nn, H2, 0);                   // emb += b2
  };

  // ---- drug entity ----
  gemm(xd, xd, Fd, W1d, nullptr, XW1, Nd, H1, Fd, 0);       // shared x@W1d
  gcn_tail(eid,  ewd,  Ed, Nd, b1d, W2d, b2d, emb1_sim);
  gcn_tail(eifd, ewfd, Ed, Nd, b1d, W2d, b2d, emb1_feat);
  gemm(emb1_sim, emb1_feat, H2, Wfd, bfd, emb1, Nd, H2, 2 * H2, 1);  // fused concat

  // ---- disease entity ----
  gemm(xs, xs, Fs, W1s, nullptr, XW1, Ns, H1, Fs, 0);       // shared x@W1s
  gcn_tail(eis,  ews,  Es, Ns, b1s, W2s, b2s, emb2_sim);
  gcn_tail(eifs, ewfs, Es, Ns, b1s, W2s, b2s, emb2_feat);
  gemm(emb2_sim, emb2_feat, H2, Wfs, bfs, emb2, Ns, H2, 2 * H2, 1);
}
